// SURVEY_BA_2grid_house_GCN_21930103013664
// MI455X (gfx1250) — matrix operand order repaired and checked by reading, not compiled, not run on hardware
//
#include <hip/hip_runtime.h>
#include <hip/hip_bf16.h>

typedef __attribute__((ext_vector_type(16))) _Float16 v16h;
typedef __attribute__((ext_vector_type(8)))  _Float16 v8h;
typedef __attribute__((ext_vector_type(8)))  float    v8f;

// ---------------------------------------------------------------------------
// Utility kernels
// ---------------------------------------------------------------------------
__global__ void k_fill(float* __restrict__ p, float v, long count) {
  long i = (long)blockIdx.x * blockDim.x + threadIdx.x;
  if (i < count) p[i] = v;
}

__global__ void k_deg_acc(const int* __restrict__ dst, const float* __restrict__ ew,
                          float* __restrict__ deg, long E) {
  long e = (long)blockIdx.x * blockDim.x + threadIdx.x;
  if (e < E) atomicAdd(&deg[dst[e]], ew[e]);
}

__global__ void k_dinv(float* __restrict__ deg, long n) {
  long i = (long)blockIdx.x * blockDim.x + threadIdx.x;
  if (i < n) {
    float d = deg[i];               // >= 1 due to self-loop init
    deg[i] = (d > 0.f) ? rsqrtf(d) : 0.f;
  }
}

// Build a 16-half WMMA fragment from two contiguous 16-byte LDS runs
// (lowers to 2x ds_load_b128).
__device__ inline v16h frag16(const _Float16* __restrict__ p0,
                              const _Float16* __restrict__ p1) {
  v8h lo = *(const v8h*)p0;
  v8h hi = *(const v8h*)p1;
  return __builtin_shufflevector(lo, hi, 0, 1, 2, 3, 4, 5, 6, 7,
                                 8, 9, 10, 11, 12, 13, 14, 15);
}

// ---------------------------------------------------------------------------
// Generic tall-skinny GEMM via V_WMMA_F32_16X16X32_F16.
// One wave computes a 16-row x (up to 64-col) tile of Y = X @ W.
// REQUIREMENT: n % 16 == 0 (true for all call sites: 100000, 512) -> no row
// guard in the store epilogue; only the lane-uniform-per-r col guard remains,
// hoisted outside the unrolled store loop.
// A staged row-major (16x64 f16); B staged TRANSPOSED (BsT[N][K]) so each
// lane's B fragment (a column of B) is contiguous in LDS. K padded to 64.
// A fragments hoisted out of the N-tile loop (reused by all N tiles).
// ---------------------------------------------------------------------------
__global__ __launch_bounds__(32)
void k_gemm_wmma(const float* __restrict__ X, int ldx,
                 const float* __restrict__ W,               // [K][Nout] row-major
                 const float* __restrict__ bias, int do_relu,
                 float* __restrict__ Y, int ldy,
                 int n, int K, int Nout) {
  __shared__ alignas(16) _Float16 As[16 * 64];    // 2 KB, row-major [M][K]
  __shared__ alignas(16) _Float16 BsT[64 * 64];   // 8 KB, transposed [N][K]
  const int lane = threadIdx.x;      // wave32
  const int row0 = blockIdx.x * 16;

  // Stage A tile (16 x 64, zero padded)
  for (int idx = lane; idx < 16 * 64; idx += 32) {
    int r = idx >> 6, c = idx & 63;
    float v = 0.f;
    if (c < K) v = X[(long)(row0 + r) * ldx + c];
    As[idx] = (_Float16)v;
  }
  // Stage B tile transposed: BsT[c][k] = W[k][c]  (64 x 64, zero padded)
  for (int idx = lane; idx < 64 * 64; idx += 32) {
    int c = idx >> 6, kk = idx & 63;
    float v = 0.f;
    if (kk < K && c < Nout) v = W[kk * Nout + c];
    BsT[idx] = (_Float16)v;
  }
  __syncthreads();

  const int lhalf = lane & 15;
  const int hi    = lane >> 4;      // 0: lanes 0-15, 1: lanes 16-31
  const bool twoK = (K > 32);
  const int ntiles = (Nout + 15) >> 4;

  // A fragments, hoisted (16-bit A 16x32 layout, ISA 7.12.2):
  //  lanes 0-15:  halves 0..7 -> K 0..7,  halves 8..15 -> K 16..23
  //  lanes 16-31: halves 0..7 -> K 8..15, halves 8..15 -> K 24..31
  const _Float16* arow = &As[lhalf * 64];
  v16h a0 = frag16(arow + (hi ? 8 : 0), arow + (hi ? 24 : 16));
  v16h a1 = {};
  if (twoK) a1 = frag16(arow + 32 + (hi ? 8 : 0), arow + 32 + (hi ? 24 : 16));

  for (int nt = 0; nt < ntiles; ++nt) {
    const int col = nt * 16 + lhalf;
    // B fragment: column `col`, K 0..7,16..23 (lanes 0-15) / 8..15,24..31
    // (lanes 16-31), matching the A layout.
    const _Float16* bcol = &BsT[col * 64];
    v8f acc = {0.f, 0.f, 0.f, 0.f, 0.f, 0.f, 0.f, 0.f};
    {
      v16h b0 = frag16(bcol + (hi ? 8 : 0), bcol + (hi ? 24 : 16));
      acc = __builtin_amdgcn_wmma_f32_16x16x32_f16(
          false, a0, false, b0, (short)0, acc, false, false);
    }
    if (twoK) {
      v16h b1 = frag16(bcol + 32 + (hi ? 8 : 0), bcol + 32 + (hi ? 24 : 16));
      acc = __builtin_amdgcn_wmma_f32_16x16x32_f16(
          false, a1, false, b1, (short)0, acc, false, false);
    }
    // C/D layout: VGPR r -> M = r + hi*8, N = col.  Row always valid
    // (n % 16 == 0); col guard is invariant across r -> single exec mask.
    if (col < Nout) {
      float badd = bias ? bias[col] : 0.f;
#pragma unroll
      for (int r = 0; r < 8; ++r) {
        long gr = row0 + r + hi * 8;
        float v = acc[r] + badd;
        if (do_relu) v = fmaxf(v, 0.f);
        Y[gr * ldy + col] = v;
      }
    }
  }
}

// ---------------------------------------------------------------------------
// Edge scatter: agg[dst] += h[src] * (dinv[src]*ew*dinv[dst]).
// 16 lanes per edge, float4 (16B) per lane -> global_load_b128 + 4 atomics.
// ---------------------------------------------------------------------------
__global__ void k_edge_scatter(const int* __restrict__ src, const int* __restrict__ dst,
                               const float* __restrict__ ew, const float* __restrict__ dinv,
                               const float* __restrict__ h, float* __restrict__ agg, long E) {
  long idx = (long)blockIdx.x * blockDim.x + threadIdx.x;
  long e   = idx >> 4;
  if (e >= E) return;
  int cg = (int)(idx & 15) * 4;
  if (cg >= 60) return;                       // cols 60..63 are padding
  int s = src[e], d = dst[e];
  const float* hp = &h[(long)s * 64 + cg];
  __builtin_prefetch(hp, 0, 0);               // global_prefetch_b8
  float norm = dinv[s] * ew[e] * dinv[d];
  float4 hv = *(const float4*)hp;             // cg <= 56 -> cg+3 <= 59 valid
  float* ap = &agg[(long)d * 64 + cg];
  atomicAdd(ap + 0, hv.x * norm);
  atomicAdd(ap + 1, hv.y * norm);
  atomicAdd(ap + 2, hv.z * norm);
  atomicAdd(ap + 3, hv.w * norm);
}

// out = relu(agg + h*dinv^2 + bias), float4 per thread
__global__ void k_self_bias_relu(const float* __restrict__ h, const float* __restrict__ dinv,
                                 const float* __restrict__ bias, const float* __restrict__ agg,
                                 float* __restrict__ xout, long n) {
  long idx = (long)blockIdx.x * blockDim.x + threadIdx.x;
  if (idx >= n * 16) return;
  int cg = (int)(idx & 15) * 4;
  if (cg >= 60) return;
  long i = idx >> 4;
  float di = dinv[i];
  float d2 = di * di;
  long off = i * 64 + cg;
  float4 av = *(const float4*)&agg[off];
  float4 hv = *(const float4*)&h[off];
  float4 bv = *(const float4*)&bias[cg];      // cg <= 56, 60 floats valid
  float4 o;
  o.x = fmaxf(av.x + hv.x * d2 + bv.x, 0.f);
  o.y = fmaxf(av.y + hv.y * d2 + bv.y, 0.f);
  o.z = fmaxf(av.z + hv.z * d2 + bv.z, 0.f);
  o.w = fmaxf(av.w + hv.w * d2 + bv.w, 0.f);
  *(float4*)&xout[off] = o;
}

// segment_max of relu'd (>=0) features via monotone uint bit-pattern atomicMax
__global__ void k_segmax(const float* __restrict__ x, const int* __restrict__ batch,
                         float* __restrict__ g, long n) {
  long idx = (long)blockIdx.x * blockDim.x + threadIdx.x;
  if (idx >= n * 16) return;
  int cg = (int)(idx & 15) * 4;
  if (cg >= 60) return;
  long i = idx >> 4;
  float4 xv = *(const float4*)&x[i * 64 + cg];
  unsigned int* gp = (unsigned int*)&g[(long)batch[i] * 64 + cg];
  atomicMax(gp + 0, __float_as_uint(xv.x));   // x >= 0: uint order == float order
  atomicMax(gp + 1, __float_as_uint(xv.y));
  atomicMax(gp + 2, __float_as_uint(xv.z));
  atomicMax(gp + 3, __float_as_uint(xv.w));
}

// ---------------------------------------------------------------------------
extern "C" void kernel_launch(void* const* d_in, const int* in_sizes, int n_in,
                              void* d_out, int out_size, void* d_ws, size_t ws_size,
                              hipStream_t stream) {
  const float* x    = (const float*)d_in[0];
  const int*   ei   = (const int*)d_in[1];
  const int*   bat  = (const int*)d_in[2];
  const float* ew   = (const float*)d_in[3];
  const float* W1 = (const float*)d_in[4];  const float* b1 = (const float*)d_in[5];
  const float* W2 = (const float*)d_in[6];  const float* b2 = (const float*)d_in[7];
  const float* W3 = (const float*)d_in[8];  const float* b3 = (const float*)d_in[9];
  const float* W4 = (const float*)d_in[10]; const float* b4 = (const float*)d_in[11];
  const float* L1w = (const float*)d_in[12]; const float* L1b = (const float*)d_in[13];
  const float* L2w = (const float*)d_in[14]; const float* L2b = (const float*)d_in[15];
  const float* L3w = (const float*)d_in[16]; const float* L3b = (const float*)d_in[17];
  float* out = (float*)d_out;

  const long N = in_sizes[0] / 10;    // 100000 nodes (multiple of 16)
  const long E = in_sizes[3];         // 1200000 edges
  const int  G = out_size / 2;        // 512 graphs (multiple of 16)
  const int* src = ei;
  const int* dst = ei + E;

  // Workspace carve-up (floats)
  float* ws   = (float*)d_ws;
  float* dinv = ws;                         // N (deg -> dinv in place)
  float* xbuf = ws + ((N + 63) & ~63L);     // N*64
  float* hbuf = xbuf + N * 64;              // N*64
  float* agg  = hbuf + N * 64;              // N*64 (head reused for pooled g0/g1/g2)
  float* g0 = agg;
  float* g1 = g0 + (long)G * 64;
  float* g2 = g1 + (long)G * 64;

  const int T = 256;
  auto blocks = [&](long cnt) { return (unsigned)((cnt + T - 1) / T); };

  // Degree + normalization: deg = 1 + segment_sum(ew, dst); dinv = rsqrt(deg)
  k_fill<<<blocks(N), T, 0, stream>>>(dinv, 1.0f, N);
  k_deg_acc<<<blocks(E), T, 0, stream>>>(dst, ew, dinv, E);
  k_dinv<<<blocks(N), T, 0, stream>>>(dinv, N);

  // 4 GCN layers
  const float* Xin = x;  int ldx = 10;  int K = 10;
  const float* Ws[4] = {W1, W2, W3, W4};
  const float* bs[4] = {b1, b2, b3, b4};
  for (int l = 0; l < 4; ++l) {
    k_gemm_wmma<<<(unsigned)((N + 15) / 16), 32, 0, stream>>>(
        Xin, ldx, Ws[l], nullptr, 0, hbuf, 64, (int)N, K, 60);
    k_fill<<<blocks(N * 64), T, 0, stream>>>(agg, 0.0f, N * 64);
    k_edge_scatter<<<blocks(E * 16), T, 0, stream>>>(src, dst, ew, dinv, hbuf, agg, E);
    k_self_bias_relu<<<blocks(N * 16), T, 0, stream>>>(hbuf, dinv, bs[l], agg, xbuf, N);
    Xin = xbuf; ldx = 64; K = 60;
  }

  // Graph max-pool (relu'd features >= 0; g0 zero-init matches empty-graph guard)
  k_fill<<<blocks((long)G * 64), T, 0, stream>>>(g0, 0.0f, (long)G * 64);
  k_segmax<<<blocks(N * 16), T, 0, stream>>>(xbuf, bat, g0, N);

  // MLP head: 60 -> 60 (relu) -> 10 (relu) -> 2
  k_gemm_wmma<<<(unsigned)((G + 15) / 16), 32, 0, stream>>>(g0, 64, L1w, L1b, 1, g1, 64, G, 60, 60);
  k_gemm_wmma<<<(unsigned)((G + 15) / 16), 32, 0, stream>>>(g1, 64, L2w, L2b, 1, g2, 64, G, 60, 10);
  k_gemm_wmma<<<(unsigned)((G + 15) / 16), 32, 0, stream>>>(g2, 64, L3w, L3b, 0, out, 2, G, 10, 2);
}
